// GAU_28037546508518
// MI455X (gfx1250) — matrix operand order repaired and checked by reading, not compiled, not run on hardware
//
#include <hip/hip_runtime.h>
#include <hip/hip_bf16.h>
#include <math.h>
#include <stdint.h>

// ---------------------------------------------------------------------------
// GAU block on gfx1250 (wave32, WMMA):
//   - all GEMMs: v_wmma_f32_16x16x32_bf16, wave tile 32Mx64N, block tile 128x128
//   - B-panels staged block-wide in LDS via global_load_async_to_lds_b128
//     (ASYNCcnt-tracked, double buffered, s_wait_asynccnt + barrier)
//   - B fragments software-pipelined depth-1 out of LDS: ds_load of b[t+1]
//     issued before the WMMAs consuming b[t]
//   - A fragments via global_load_b128, software-pipelined by the unroller
// Pipeline: prep weights -> LN -> gemm_h(SiLU,split v/gate) -> gemm_qk ->
//           per-batch qk^T/S -> masked softmax -> per-batch A@v * gate ->
//           @Wo + bo + residual.
// ---------------------------------------------------------------------------

typedef __attribute__((ext_vector_type(16))) __bf16 v16bf;
typedef __attribute__((ext_vector_type(8)))  float  v8f;

#define BB   8
#define SS   2048
#define DD   512
#define QKD  128
#define HIDN 1024
#define NH   2048
#define NQK  256
#define MTOT (BB*SS)

#define LDSROW     80                  // 64B of K-data + 16B pad (bank-conflict-free)
#define TILEB_SZ   (128 * LDSROW)      // one 128N x 32K bf16 panel
#define SMEM_SZ    (2 * TILEB_SZ)      // double buffered

__device__ __forceinline__ float silu_f(float x) { return x / (1.0f + __expf(-x)); }

// ---- WMMA fragment loaders -------------------------------------------------
// A 16x32 bf16 (ISA 7.12.2): lane = m + 16*half;
//   half=0: elems 0..7 = K0..7,  8..15 = K16..23
//   half=1: elems 0..7 = K8..15, 8..15 = K24..31
__device__ __forceinline__ v16bf load_fragA(const __bf16* base, int ld, int kb) {
  int lane = threadIdx.x & 31;
  int half = lane >> 4, r = lane & 15;
  const __bf16* p = base + (size_t)r * ld + kb + half * 8;
  union { v16bf v; float4 q[2]; } u;
  u.q[0] = *(const float4*)(const void*)(p);
  u.q[1] = *(const float4*)(const void*)(p + 16);
  return u.v;
}

// B 32x16 bf16: lane = n + 16*half; elems 0..7 = K half*8..+7, 8..15 = K 16+half*8..+7.
// Panel in LDS is N-major with LDSROW-byte row stride.
__device__ __forceinline__ v16bf load_fragB_lds(const char* sm, int nloc) {
  int lane = threadIdx.x & 31;
  int half = lane >> 4, r = lane & 15;
  const char* p = sm + (size_t)(nloc + r) * LDSROW + half * 16;
  union { v16bf v; float4 q[2]; } u;
  u.q[0] = *(const float4*)(const void*)(p);
  u.q[1] = *(const float4*)(const void*)(p + 32);
  return u.v;
}

// ---- async B-panel stage: 128 rows x 64B, 512 x 16B chunks, 2 per thread ---
__device__ __forceinline__ void issue_panel(char* smbuf, const __bf16* __restrict__ Bblk,
                                            int ldb, int kb) {
  int tid = threadIdx.x;
#pragma unroll
  for (int i = 0; i < 2; ++i) {
    int c   = tid + i * 256;           // 0..511
    int row = c >> 2;                  // 0..127
    int off = c & 3;                   // 16B chunk within row
    unsigned lds = (unsigned)(uintptr_t)(smbuf + (size_t)row * LDSROW + off * 16);
    unsigned long long g =
        (unsigned long long)(uintptr_t)(Bblk + (size_t)row * ldb + kb + off * 8);
    asm volatile("global_load_async_to_lds_b128 %0, %1, off"
                 :: "v"(lds), "v"(g) : "memory");
  }
}

__device__ __forceinline__ void wait_async_all() {
  asm volatile("s_wait_asynccnt 0x0" ::: "memory");
}

// ---- block GEMM core: 128M x 128N tile, wave = 32M x 64N -------------------
__device__ __forceinline__ void gemm_core(const __bf16* __restrict__ Ablk, int lda,
                                          const __bf16* __restrict__ Bblk, int ldb,
                                          int K, char* sm, v8f acc[2][4]) {
  int wid    = threadIdx.x >> 5;
  int wave_m = (wid & 3) * 32;
  int wave_n = (wid >> 2) * 64;
  const __bf16* Arow = Ablk + (size_t)wave_m * lda;

  issue_panel(sm, Bblk, ldb, 0);
  int buf = 0;
  for (int kb = 0; kb < K; kb += 32, buf ^= 1) {
    // A fragments from global — issued before the async wait to overlap.
    v16bf a0 = load_fragA(Arow,                    lda, kb);
    v16bf a1 = load_fragA(Arow + (size_t)16 * lda, lda, kb);
    wait_async_all();
    __syncthreads();                       // panel `buf` visible block-wide
    if (kb + 32 < K) issue_panel(sm + (buf ^ 1) * TILEB_SZ, Bblk, ldb, kb + 32);
    const char* smb = sm + buf * TILEB_SZ;
    // Depth-1 pipeline on B fragments: ds_load of b[t+1] precedes WMMAs of b[t].
    v16bf bcur = load_fragB_lds(smb, wave_n);
#pragma unroll
    for (int t = 0; t < 4; ++t) {
      v16bf bnext;
      if (t < 3) bnext = load_fragB_lds(smb, wave_n + (t + 1) * 16);
      acc[0][t] = __builtin_amdgcn_wmma_f32_16x16x32_bf16(
          false, a0, false, bcur, (short)0, acc[0][t], false, false);
      acc[1][t] = __builtin_amdgcn_wmma_f32_16x16x32_bf16(
          false, a1, false, bcur, (short)0, acc[1][t], false, false);
      bcur = bnext;
    }
    __syncthreads();                       // reads of `buf` done before rewrite
  }
}

__device__ __forceinline__ void epi_coords(int& mbase, int& nbase) {
  int wid  = threadIdx.x >> 5;
  int lane = threadIdx.x & 31;
  mbase = blockIdx.y * 128 + (wid & 3) * 32 + (lane >> 4) * 8;
  nbase = blockIdx.x * 128 + (wid >> 2) * 64 + (lane & 15);
}

// ---------------------------------------------------------------------------
__global__ void k_transpose_cast(const float* __restrict__ src, __bf16* __restrict__ dst,
                                 int K, int N) {  // dst[n][k] = src[k][n]
  size_t idx = (size_t)blockIdx.x * blockDim.x + threadIdx.x;
  size_t total = (size_t)K * N;
  if (idx >= total) return;
  size_t n = idx / (size_t)K, k = idx - n * (size_t)K;
  dst[idx] = (__bf16)src[k * (size_t)N + n];
}

// One wave32 per row of 512: 16 contiguous floats/lane, shuffle reduce.
__global__ void k_layernorm(const float* __restrict__ x, const float* __restrict__ g,
                            const float* __restrict__ b, __bf16* __restrict__ nx) {
  int row  = blockIdx.x * 8 + (threadIdx.x >> 5);
  int lane = threadIdx.x & 31;
  const float* xr = x + (size_t)row * DD + lane * 16;
  float vals[16];
#pragma unroll
  for (int c = 0; c < 4; ++c) {
    float4 v = *(const float4*)(const void*)(xr + c * 4);
    vals[c * 4 + 0] = v.x; vals[c * 4 + 1] = v.y;
    vals[c * 4 + 2] = v.z; vals[c * 4 + 3] = v.w;
  }
  float s = 0.0f, s2 = 0.0f;
#pragma unroll
  for (int i = 0; i < 16; ++i) { s += vals[i]; s2 += vals[i] * vals[i]; }
  for (int off = 16; off > 0; off >>= 1) {
    s  += __shfl_xor(s,  off, 32);
    s2 += __shfl_xor(s2, off, 32);
  }
  float mu   = s * (1.0f / DD);
  float var  = s2 * (1.0f / DD) - mu * mu;
  float rstd = rsqrtf(var + 1e-5f);
  const float* gp = g + lane * 16;
  const float* bp = b + lane * 16;
  union { __bf16 h[16]; float4 q[2]; } o;
#pragma unroll
  for (int i = 0; i < 16; ++i)
    o.h[i] = (__bf16)((vals[i] - mu) * rstd * gp[i] + bp[i]);
  __bf16* np = nx + (size_t)row * DD + lane * 16;
  *(float4*)(void*)(np)     = o.q[0];
  *(float4*)(void*)(np + 8) = o.q[1];
}

// h = silu(nx @ Wh + bh); n<1024 -> vT[b][n][s], else gate[m][n-1024]
__global__ void k_gemm_h(const __bf16* __restrict__ A, const __bf16* __restrict__ Bt,
                         const float* __restrict__ bias,
                         __bf16* __restrict__ vT, __bf16* __restrict__ gate) {
  __shared__ __align__(16) char sm[SMEM_SZ];
  v8f acc[2][4] = {};
  gemm_core(A + (size_t)blockIdx.y * 128 * DD, DD,
            Bt + (size_t)blockIdx.x * 128 * DD, DD, DD, sm, acc);
  int mbase, nbase; epi_coords(mbase, nbase);
#pragma unroll
  for (int gI = 0; gI < 2; ++gI)
#pragma unroll
    for (int t = 0; t < 4; ++t)
#pragma unroll
      for (int r = 0; r < 8; ++r) {
        int m = mbase + gI * 16 + r, n = nbase + t * 16;
        float val = silu_f(acc[gI][t][r] + bias[n]);
        if (n < HIDN) {
          int bI = m >> 11, s = m & (SS - 1);
          vT[((size_t)bI * HIDN + n) * SS + s] = (__bf16)val;
        } else {
          gate[(size_t)m * HIDN + (n - HIDN)] = (__bf16)val;
        }
      }
}

// qk = silu(nx @ Wqk + bqk); n<128 -> q, else k
__global__ void k_gemm_qk(const __bf16* __restrict__ A, const __bf16* __restrict__ Bt,
                          const float* __restrict__ bias,
                          __bf16* __restrict__ q, __bf16* __restrict__ k) {
  __shared__ __align__(16) char sm[SMEM_SZ];
  v8f acc[2][4] = {};
  gemm_core(A + (size_t)blockIdx.y * 128 * DD, DD,
            Bt + (size_t)blockIdx.x * 128 * DD, DD, DD, sm, acc);
  int mbase, nbase; epi_coords(mbase, nbase);
#pragma unroll
  for (int gI = 0; gI < 2; ++gI)
#pragma unroll
    for (int t = 0; t < 4; ++t)
#pragma unroll
      for (int r = 0; r < 8; ++r) {
        int m = mbase + gI * 16 + r, n = nbase + t * 16;
        float val = silu_f(acc[gI][t][r] + bias[n]);
        if (n < QKD) q[(size_t)m * QKD + n] = (__bf16)val;
        else         k[(size_t)m * QKD + (n - QKD)] = (__bf16)val;
      }
}

// sim[b][i][j] = (q_i . k_j) / S  (per-batch, K=128)
__global__ void k_gemm_sim(const __bf16* __restrict__ q, const __bf16* __restrict__ k,
                           float* __restrict__ sim) {
  __shared__ __align__(16) char sm[SMEM_SZ];
  int bI = blockIdx.z;
  v8f acc[2][4] = {};
  gemm_core(q + (size_t)bI * SS * QKD + (size_t)blockIdx.y * 128 * QKD, QKD,
            k + (size_t)bI * SS * QKD + (size_t)blockIdx.x * 128 * QKD, QKD,
            QKD, sm, acc);
  int mbase, nbase; epi_coords(mbase, nbase);
  float* simb = sim + (size_t)bI * SS * SS;
  const float scale = 1.0f / (float)SS;
#pragma unroll
  for (int gI = 0; gI < 2; ++gI)
#pragma unroll
    for (int t = 0; t < 4; ++t)
#pragma unroll
      for (int r = 0; r < 8; ++r)
        simb[(size_t)(mbase + gI * 16 + r) * SS + nbase + t * 16] =
            acc[gI][t][r] * scale;
}

// Row softmax with attention mask; bf16 probabilities (WMMA A layout = row-major).
__global__ void k_softmax(const float* __restrict__ sim, const int* __restrict__ mask,
                          __bf16* __restrict__ probs) {
  int row = blockIdx.x;
  int bI  = row >> 11;
  const float* sr = sim + (size_t)row * SS;
  const int*   mr = mask + (size_t)bI * SS;
  int tid = threadIdx.x, lane = tid & 31, wid = tid >> 5;
  float vals[8], mx = -3.0e38f;
#pragma unroll
  for (int i = 0; i < 8; ++i) {
    int j = tid + i * 256;
    float v = sr[j];
    if (mr[j] == 0) v = -1.0e9f;
    vals[i] = v; mx = fmaxf(mx, v);
  }
  __shared__ float red[8];
  for (int off = 16; off > 0; off >>= 1) mx = fmaxf(mx, __shfl_xor(mx, off, 32));
  if (lane == 0) red[wid] = mx;
  __syncthreads();
  mx = red[0];
#pragma unroll
  for (int w = 1; w < 8; ++w) mx = fmaxf(mx, red[w]);
  __syncthreads();
  float sum = 0.0f;
#pragma unroll
  for (int i = 0; i < 8; ++i) { float e = __expf(vals[i] - mx); vals[i] = e; sum += e; }
  for (int off = 16; off > 0; off >>= 1) sum += __shfl_xor(sum, off, 32);
  if (lane == 0) red[wid] = sum;
  __syncthreads();
  sum = red[0];
#pragma unroll
  for (int w = 1; w < 8; ++w) sum += red[w];
  float inv = 1.0f / sum;
#pragma unroll
  for (int i = 0; i < 8; ++i) {
    int j = tid + i * 256;
    probs[(size_t)row * SS + j] = (__bf16)(vals[i] * inv);
  }
}

// avg[m][n] = (probs @ v)[i][n] * gate[m][n]  (per-batch, K=2048)
__global__ void k_gemm_av(const __bf16* __restrict__ probs, const __bf16* __restrict__ vT,
                          const __bf16* __restrict__ gate, __bf16* __restrict__ avg) {
  __shared__ __align__(16) char sm[SMEM_SZ];
  int bI = blockIdx.z;
  v8f acc[2][4] = {};
  gemm_core(probs + (size_t)bI * SS * SS + (size_t)blockIdx.y * 128 * SS, SS,
            vT + (size_t)bI * HIDN * SS + (size_t)blockIdx.x * 128 * SS, SS,
            SS, sm, acc);
  int mbase, nbase; epi_coords(mbase, nbase);
#pragma unroll
  for (int gI = 0; gI < 2; ++gI)
#pragma unroll
    for (int t = 0; t < 4; ++t)
#pragma unroll
      for (int r = 0; r < 8; ++r) {
        int gm = bI * SS + mbase + gI * 16 + r;
        size_t idx = (size_t)gm * HIDN + nbase + t * 16;
        avg[idx] = (__bf16)(acc[gI][t][r] * (float)gate[idx]);
      }
}

// out = avg @ Wo + bo + x  (K=1024, f32 out)
__global__ void k_gemm_out(const __bf16* __restrict__ A, const __bf16* __restrict__ Bt,
                           const float* __restrict__ bias, const float* __restrict__ x,
                           float* __restrict__ out) {
  __shared__ __align__(16) char sm[SMEM_SZ];
  v8f acc[2][4] = {};
  gemm_core(A + (size_t)blockIdx.y * 128 * HIDN, HIDN,
            Bt + (size_t)blockIdx.x * 128 * HIDN, HIDN, HIDN, sm, acc);
  int mbase, nbase; epi_coords(mbase, nbase);
#pragma unroll
  for (int gI = 0; gI < 2; ++gI)
#pragma unroll
    for (int t = 0; t < 4; ++t)
#pragma unroll
      for (int r = 0; r < 8; ++r) {
        int m = mbase + gI * 16 + r, n = nbase + t * 16;
        size_t idx = (size_t)m * DD + n;
        out[idx] = acc[gI][t][r] + bias[n] + x[idx];
      }
}

// ---------------------------------------------------------------------------
extern "C" void kernel_launch(void* const* d_in, const int* in_sizes, int n_in,
                              void* d_out, int out_size, void* d_ws, size_t ws_size,
                              hipStream_t stream) {
  const float* x    = (const float*)d_in[0];
  const int*   mask = (const int*)d_in[1];
  const float* ln_g = (const float*)d_in[2];
  const float* ln_b = (const float*)d_in[3];
  const float* Wh   = (const float*)d_in[4];
  const float* bh   = (const float*)d_in[5];
  const float* Wqk  = (const float*)d_in[6];
  const float* bqk  = (const float*)d_in[7];
  const float* Wo   = (const float*)d_in[8];
  const float* bo   = (const float*)d_in[9];
  float* out = (float*)d_out;

  char* p = (char*)d_ws;
  auto carve = [&](size_t bytes) -> char* {
    char* r = p; p += (bytes + 255) & ~(size_t)255; return r;
  };
  __bf16* nx    = (__bf16*)carve((size_t)MTOT * DD * 2);
  __bf16* WhT   = (__bf16*)carve((size_t)NH * DD * 2);
  __bf16* WqkT  = (__bf16*)carve((size_t)NQK * DD * 2);
  __bf16* WoT   = (__bf16*)carve((size_t)DD * HIDN * 2);
  __bf16* vT    = (__bf16*)carve((size_t)BB * HIDN * SS * 2);
  __bf16* gate  = (__bf16*)carve((size_t)MTOT * HIDN * 2);
  __bf16* qbuf  = (__bf16*)carve((size_t)MTOT * QKD * 2);
  __bf16* kbuf  = (__bf16*)carve((size_t)MTOT * QKD * 2);
  float*  sim   = (float*) carve((size_t)BB * SS * SS * 4);
  __bf16* probs = (__bf16*)carve((size_t)BB * SS * SS * 2);
  __bf16* avg   = (__bf16*)carve((size_t)MTOT * HIDN * 2);

  {
    size_t t1 = (size_t)DD * NH, t2 = (size_t)DD * NQK, t3 = (size_t)HIDN * DD;
    k_transpose_cast<<<dim3((unsigned)((t1 + 255) / 256)), 256, 0, stream>>>(Wh,  WhT,  DD,   NH);
    k_transpose_cast<<<dim3((unsigned)((t2 + 255) / 256)), 256, 0, stream>>>(Wqk, WqkT, DD,   NQK);
    k_transpose_cast<<<dim3((unsigned)((t3 + 255) / 256)), 256, 0, stream>>>(Wo,  WoT,  HIDN, DD);
  }

  k_layernorm<<<dim3(MTOT / 8), 256, 0, stream>>>(x, ln_g, ln_b, nx);

  k_gemm_h  <<<dim3(NH / 128,  MTOT / 128), 256, 0, stream>>>(nx, WhT,  bh,  vT, gate);
  k_gemm_qk <<<dim3(NQK / 128, MTOT / 128), 256, 0, stream>>>(nx, WqkT, bqk, qbuf, kbuf);
  k_gemm_sim<<<dim3(SS / 128,  SS / 128, BB), 256, 0, stream>>>(qbuf, kbuf, sim);
  k_softmax <<<dim3(MTOT), 256, 0, stream>>>(sim, mask, probs);
  k_gemm_av <<<dim3(HIDN / 128, SS / 128, BB), 256, 0, stream>>>(probs, vT, gate, avg);
  k_gemm_out<<<dim3(DD / 128,  MTOT / 128), 256, 0, stream>>>(avg, WoT, bo, x, out);
}
